// FlaGdnpCompatibleSequenceMixer_47064251630313
// MI455X (gfx1250) — matrix operand order repaired and checked by reading, not compiled, not run on hardware
//
#include <hip/hip_runtime.h>
#include <hip/hip_bf16.h>
#include <stdint.h>

typedef __attribute__((ext_vector_type(16))) __bf16 v16bf;
typedef __attribute__((ext_vector_type(8)))  float  v8f;
typedef int v4i __attribute__((vector_size(16)));   // matches builtin param type

#define BDIM    2
#define SLEN    4096
#define DMODEL  1024
#define NHEADS  16
#define HDIM    64
#define MROWS   (BDIM * SLEN)                       /* 8192 */
#define CTRLN   (3 * DMODEL + 2 * NHEADS + DMODEL)  /* 4128 */
#define P3N     (3 * DMODEL)                        /* 3072 */
#define FUSEDK  (2 * DMODEL)                        /* 2048 */

#ifndef __has_builtin
#define __has_builtin(x) 0
#endif
#if __has_builtin(__builtin_amdgcn_global_load_async_to_lds_b128)
#define HAVE_ASYNC_LDS 1
#else
#define HAVE_ASYNC_LDS 0
#endif

// 16-byte global->LDS copy: async DMA (ASYNCcnt) when available, else via regs.
__device__ __forceinline__ void cp16_g2l(const void* g, void* l) {
#if HAVE_ASYNC_LDS
  __builtin_amdgcn_global_load_async_to_lds_b128((v4i*)g, (v4i*)l, 0, 0);
#else
  *(uint4*)l = *(const uint4*)g;
#endif
}

__device__ __forceinline__ void cp_wait() {
#if HAVE_ASYNC_LDS
#if __has_builtin(__builtin_amdgcn_s_wait_asynccnt)
  __builtin_amdgcn_s_wait_asynccnt(0);
#else
  asm volatile("s_wait_asynccnt 0" ::: "memory");
#endif
#endif
}

__device__ __forceinline__ float sigmoidf_(float x) { return 1.0f / (1.0f + __expf(-x)); }
__device__ __forceinline__ float siluf_(float x)    { return x * sigmoidf_(x); }

// ---------------------------------------------------------------- f32 -> bf16
__global__ __launch_bounds__(256) void k_f2bf(const float* __restrict__ in,
                                              __bf16* __restrict__ out, int n) {
  int i = blockIdx.x * 256 + threadIdx.x;
  if (i < n) out[i] = (__bf16)in[i];
}

// ------------------- pack weights W[K,N] f32 -> W'[K/32][N][32] bf16
// (k-contiguous per n-row so GEMM B-staging is a straight aligned copy)
__global__ __launch_bounds__(256) void k_packB(const float* __restrict__ in,
                                               __bf16* __restrict__ out,
                                               int K, int N) {
  int idx = blockIdx.x * 256 + threadIdx.x;
  if (idx >= K * N) return;
  int n = idx % N;
  int k = idx / N;
  out[((size_t)(k >> 5) * N + n) * 32 + (k & 31)] = (__bf16)in[idx];
}

// ------------------------------------------------- P20 complex rotary scan
__global__ __launch_bounds__(256) void k_p20_scan(const float* __restrict__ hidden,
                                                  const float* __restrict__ dlog,
                                                  const float* __restrict__ theta,
                                                  float* __restrict__ p20) {
  int gid = blockIdx.x * 256 + threadIdx.x;           // 0 .. 1023
  int b = gid / (DMODEL / 2);
  int p = gid % (DMODEL / 2);
  float mag = __expf(-__expf(dlog[p]));
  float ar = mag * __cosf(theta[p]);
  float ai = mag * __sinf(theta[p]);
  float sr = 0.f, si = 0.f;
  const float* xb = hidden + (size_t)b * SLEN * DMODEL + 2 * p;
  float* ob = p20 + (size_t)b * SLEN * DMODEL + 2 * p;
  for (int t = 0; t < SLEN; ++t) {
    float2 u = *(const float2*)(xb + (size_t)t * DMODEL);
    float nr = ar * sr - ai * si + u.x;
    float ni = ar * si + ai * sr + u.y;
    sr = nr; si = ni;
    *(float2*)(ob + (size_t)t * DMODEL) = make_float2(sr, si);
  }
}

// ------------------------------------------------------------- WMMA bf16 GEMM
// C[M,N] = A[M,K](bf16, lda) @ Bp (prepacked [K/32][N][32] bf16)
//          (+bias) (* sigmoid(gate)), fp32 out.
// Block tile 128x128, BK=32, 256 threads = 8 waves (2x4), wave tile 64x32.
// Double-buffered LDS; async global->LDS staging when the toolchain has it.
__global__ __launch_bounds__(256) void k_gemm_bf16(
    const __bf16* __restrict__ A, int lda,
    const __bf16* __restrict__ Bp, int N, int K,
    float* __restrict__ C,
    const float* __restrict__ bias,
    const float* __restrict__ gate, int gate_stride) {
  __shared__ __align__(16) __bf16 As[2][128 * 40];
  __shared__ __align__(16) __bf16 Bs[2][128 * 40];

  int tid = threadIdx.x;
  int lane = tid & 31, wid = tid >> 5;
  int waveM = wid >> 2;          // 0..1
  int waveN = wid & 3;           // 0..3
  int lhalf = lane >> 4;         // 0/1
  int l16 = lane & 15;

  int m0 = blockIdx.y * 128;
  int n0 = blockIdx.x * 128;

  v8f zero;
#pragma unroll
  for (int i = 0; i < 8; ++i) zero[i] = 0.f;
  v8f acc[4][2];
#pragma unroll
  for (int mi = 0; mi < 4; ++mi)
#pragma unroll
    for (int ni = 0; ni < 2; ++ni) acc[mi][ni] = zero;

  // global->LDS mappings (loop-invariant): row 0..127, 16-half chunk 0..1
  int row = tid >> 1, chk = tid & 1;
  int nbc = n0 + row;
  if (nbc > N - 1) nbc = N - 1;                 // clamp; dead cols never stored
  const __bf16* pa_base = A + (size_t)(m0 + row) * lda + chk * 16;
  const __bf16* pb_base = Bp + (size_t)nbc * 32 + chk * 16;
  __bf16* la = &As[0][row * 40 + chk * 16];
  __bf16* lb = &Bs[0][row * 40 + chk * 16];
  const int lbuf = 128 * 40;                    // halves per buffer

  auto stage = [&](int buf, int k0) {
    const __bf16* pa = pa_base + k0;
    cp16_g2l(pa,     la + buf * lbuf);
    cp16_g2l(pa + 8, la + buf * lbuf + 8);
    const __bf16* pb = pb_base + (size_t)(k0 >> 5) * N * 32;
    cp16_g2l(pb,     lb + buf * lbuf);
    cp16_g2l(pb + 8, lb + buf * lbuf + 8);
  };

  stage(0, 0);
  cp_wait();
  __syncthreads();

  int nsteps = K >> 5;
  for (int s = 0; s < nsteps; ++s) {
    int cur = s & 1;
    if (s + 1 < nsteps) stage(cur ^ 1, (s + 1) << 5);

    union { uint4 q[2]; v16bf v; } af[4], bfr[2];
    const __bf16* abase = &As[cur][(waveM * 64 + l16) * 40 + lhalf * 16];
    const __bf16* bbase = &Bs[cur][(waveN * 32 + l16) * 40 + lhalf * 16];
#pragma unroll
    for (int mi = 0; mi < 4; ++mi) {
      const __bf16* p = abase + mi * 16 * 40;
      af[mi].q[0] = *(const uint4*)p;
      af[mi].q[1] = *(const uint4*)(p + 8);
    }
#pragma unroll
    for (int ni = 0; ni < 2; ++ni) {
      const __bf16* p = bbase + ni * 16 * 40;
      bfr[ni].q[0] = *(const uint4*)p;
      bfr[ni].q[1] = *(const uint4*)(p + 8);
    }
#pragma unroll
    for (int mi = 0; mi < 4; ++mi)
#pragma unroll
      for (int ni = 0; ni < 2; ++ni)
        acc[mi][ni] = __builtin_amdgcn_wmma_f32_16x16x32_bf16(
            false, af[mi].v, false, bfr[ni].v, (short)0, acc[mi][ni], false, false);
    cp_wait();
    __syncthreads();
  }

  // epilogue: bias, optional gate multiply, fp32 store
#pragma unroll
  for (int mi = 0; mi < 4; ++mi) {
#pragma unroll
    for (int ni = 0; ni < 2; ++ni) {
      int n = n0 + waveN * 32 + ni * 16 + l16;
      if (n >= N) continue;
      float bv = bias ? bias[n] : 0.f;
#pragma unroll
      for (int v = 0; v < 8; ++v) {
        int m = m0 + waveM * 64 + mi * 16 + lhalf * 8 + v;
        float val = acc[mi][ni][v] + bv;
        if (gate) val *= sigmoidf_(gate[(size_t)m * gate_stride + n]);
        C[(size_t)m * N + n] = val;
      }
    }
  }
}

// --------------------------------------------- RMSNorm(p20) -> fused[:,1024:]
__global__ __launch_bounds__(256) void k_p20norm(const float* __restrict__ p20,
                                                 const float* __restrict__ vscale,
                                                 __bf16* __restrict__ fused) {
  __shared__ float part[8];
  int bt = blockIdx.x;               // 0..MROWS-1
  int tid = threadIdx.x;
  const float* row = p20 + (size_t)bt * DMODEL;
  float x[4];
  float ss = 0.f;
#pragma unroll
  for (int j = 0; j < 4; ++j) {
    x[j] = row[tid + j * 256];
    ss += x[j] * x[j];
  }
#pragma unroll
  for (int o = 16; o > 0; o >>= 1) ss += __shfl_xor(ss, o);
  if ((tid & 31) == 0) part[tid >> 5] = ss;
  __syncthreads();
  float tot = 0.f;
#pragma unroll
  for (int w = 0; w < 8; ++w) tot += part[w];
  float r = rsqrtf(tot / DMODEL + 1e-6f);
  __bf16* out = fused + (size_t)bt * FUSEDK + DMODEL;
#pragma unroll
  for (int j = 0; j < 4; ++j) {
    int d = tid + j * 256;
    out[d] = (__bf16)(x[j] * r * vscale[d]);
  }
}

// --------------------------------------------------- g = logsigmoid, beta = sigmoid
__global__ __launch_bounds__(256) void k_gbeta(const float* __restrict__ ctrl,
                                               float* __restrict__ g,
                                               float* __restrict__ beta) {
  int i = blockIdx.x * 256 + threadIdx.x;      // over B*S*H
  if (i >= MROWS * NHEADS) return;
  int h = i % NHEADS;
  int bt = i / NHEADS;
  float dec = ctrl[(size_t)bt * CTRLN + 3 * DMODEL + h];
  float bin = ctrl[(size_t)bt * CTRLN + 3 * DMODEL + NHEADS + h];
  float gs = fminf(dec, 0.f) - log1pf(__expf(-fabsf(dec)));   // stable logsigmoid
  g[i] = gs;
  beta[i] = sigmoidf_(bin);
}

// ------------------------- causal depthwise conv(K=4) + SiLU + ramp * p20 branch
__global__ __launch_bounds__(256) void k_conv(const float* __restrict__ ctrl,
                                              const float* __restrict__ w,
                                              const float* __restrict__ ramp_logit,
                                              const float* __restrict__ p20qkv,
                                              float* __restrict__ out, int sel) {
  int i = blockIdx.x * 256 + threadIdx.x;      // over B*S*D
  int d = i & (DMODEL - 1);
  int t = (i >> 10) & (SLEN - 1);
  int b = i >> 22;
  int bt = b * SLEN + t;
  float acc = 0.f;
#pragma unroll
  for (int j = 0; j < 4; ++j) {
    int tt = t - 3 + j;
    if (tt >= 0)
      acc += ctrl[(size_t)(b * SLEN + tt) * CTRLN + sel * DMODEL + d] * w[d * 4 + j];
  }
  float conv = siluf_(acc);
  float ramp = sigmoidf_(ramp_logit[d]);
  float p20v = p20qkv[(size_t)bt * P3N + sel * DMODEL + d];
  out[(size_t)bt * DMODEL + d] = conv + ramp * p20v;
}

// --------------------------------------------- gated delta rule (seq scan)
// one 64-thread block per (b,h); thread v owns state column [64] in registers.
__global__ __launch_bounds__(64) void k_delta(const float* __restrict__ q,
                                              const float* __restrict__ k,
                                              const float* __restrict__ v,
                                              const float* __restrict__ g,
                                              const float* __restrict__ beta,
                                              float* __restrict__ reads) {
  __shared__ __align__(16) float sk[HDIM];
  __shared__ __align__(16) float sq[HDIM];
  __shared__ float red[4];
  int bh = blockIdx.x;
  int b = bh / NHEADS, h = bh % NHEADS;
  int tid = threadIdx.x;               // v-column
  int wid = tid >> 5;

  float st[HDIM];
#pragma unroll
  for (int j = 0; j < HDIM; ++j) st[j] = 0.f;

  for (int t = 0; t < SLEN; ++t) {
    size_t base = (size_t)(b * SLEN + t) * DMODEL + h * HDIM;
    float kv = k[base + tid];
    float qv = q[base + tid];
    float vv = v[base + tid];
    size_t gi = (size_t)(b * SLEN + t) * NHEADS + h;
    float decay = __expf(g[gi]);
    float bt = beta[gi];

    float ssk = kv * kv, ssq = qv * qv;
#pragma unroll
    for (int o = 16; o > 0; o >>= 1) {
      ssk += __shfl_xor(ssk, o);
      ssq += __shfl_xor(ssq, o);
    }
    if ((tid & 31) == 0) { red[wid] = ssk; red[2 + wid] = ssq; }
    __syncthreads();
    float rk = rsqrtf(red[0] + red[1] + 1e-6f);
    float rq = rsqrtf(red[2] + red[3] + 1e-6f);
    sk[tid] = kv * rk;
    sq[tid] = qv * rq;
    __syncthreads();

    float dot = 0.f;
#pragma unroll
    for (int j4 = 0; j4 < HDIM / 4; ++j4) {
      float4 kk = *(const float4*)&sk[4 * j4];
      st[4 * j4 + 0] *= decay; dot = fmaf(st[4 * j4 + 0], kk.x, dot);
      st[4 * j4 + 1] *= decay; dot = fmaf(st[4 * j4 + 1], kk.y, dot);
      st[4 * j4 + 2] *= decay; dot = fmaf(st[4 * j4 + 2], kk.z, dot);
      st[4 * j4 + 3] *= decay; dot = fmaf(st[4 * j4 + 3], kk.w, dot);
    }
    float err = (vv - dot) * bt;
    float qdot = 0.f;
#pragma unroll
    for (int j4 = 0; j4 < HDIM / 4; ++j4) {
      float4 kk = *(const float4*)&sk[4 * j4];
      float4 qq = *(const float4*)&sq[4 * j4];
      st[4 * j4 + 0] = fmaf(kk.x, err, st[4 * j4 + 0]); qdot = fmaf(st[4 * j4 + 0], qq.x, qdot);
      st[4 * j4 + 1] = fmaf(kk.y, err, st[4 * j4 + 1]); qdot = fmaf(st[4 * j4 + 1], qq.y, qdot);
      st[4 * j4 + 2] = fmaf(kk.z, err, st[4 * j4 + 2]); qdot = fmaf(st[4 * j4 + 2], qq.z, qdot);
      st[4 * j4 + 3] = fmaf(kk.w, err, st[4 * j4 + 3]); qdot = fmaf(st[4 * j4 + 3], qq.w, qdot);
    }
    reads[base + tid] = qdot * 0.125f;   // HD^-0.5
    __syncthreads();
  }
}

// -------------------------------- per-head RMSNorm(reads) -> fused[:,0:1024]
__global__ __launch_bounds__(256) void k_mnorm(const float* __restrict__ reads,
                                               const float* __restrict__ mscale,
                                               __bf16* __restrict__ fused) {
  int gw = blockIdx.x * 8 + (threadIdx.x >> 5);   // global (b,t,h)
  int lane = threadIdx.x & 31;
  int h = gw % NHEADS;
  int bt = gw / NHEADS;
  size_t base = (size_t)bt * DMODEL + h * HDIM;
  float x0 = reads[base + lane];
  float x1 = reads[base + lane + 32];
  float ss = x0 * x0 + x1 * x1;
#pragma unroll
  for (int o = 16; o > 0; o >>= 1) ss += __shfl_xor(ss, o);
  float r = rsqrtf(ss / HDIM + 1e-6f);
  __bf16* out = fused + (size_t)bt * FUSEDK + h * HDIM;
  out[lane] = (__bf16)(x0 * r * mscale[lane]);
  out[lane + 32] = (__bf16)(x1 * r * mscale[lane + 32]);
}

// ============================================================================
extern "C" void kernel_launch(void* const* d_in, const int* in_sizes, int n_in,
                              void* d_out, int out_size, void* d_ws, size_t ws_size,
                              hipStream_t stream) {
  const float* hidden       = (const float*)d_in[0];
  const float* p20_dlog     = (const float*)d_in[1];
  const float* p20_theta    = (const float*)d_in[2];
  const float* ctrl_W       = (const float*)d_in[3];
  const float* ctrl_b       = (const float*)d_in[4];
  const float* p20c_W       = (const float*)d_in[5];
  const float* qconv_w      = (const float*)d_in[6];
  const float* kconv_w      = (const float*)d_in[7];
  const float* vconv_w      = (const float*)d_in[8];
  const float* mnorm_scale  = (const float*)d_in[9];
  const float* vnorm_scale  = (const float*)d_in[10];
  const float* ramp_logit   = (const float*)d_in[11];
  const float* out_W        = (const float*)d_in[12];
  const float* out_b        = (const float*)d_in[13];
  float* out = (float*)d_out;

  char* ws = (char*)d_ws;
  size_t cur = 0;
  auto alloc = [&](size_t bytes) { void* p = ws + cur; cur += (bytes + 255) & ~(size_t)255; return p; };

  __bf16* hb      = (__bf16*)alloc((size_t)MROWS * DMODEL * 2);
  __bf16* ctrlWb  = (__bf16*)alloc((size_t)DMODEL * CTRLN * 2);
  __bf16* p20cWb  = (__bf16*)alloc((size_t)DMODEL * P3N * 2);
  __bf16* outWb   = (__bf16*)alloc((size_t)FUSEDK * DMODEL * 2);
  float*  ctrl    = (float*)alloc((size_t)MROWS * CTRLN * 4);
  float*  p20rd   = (float*)alloc((size_t)MROWS * DMODEL * 4);
  float*  p20qkv  = (float*)alloc((size_t)MROWS * P3N * 4);
  float*  qb      = (float*)alloc((size_t)MROWS * DMODEL * 4);
  float*  kb      = (float*)alloc((size_t)MROWS * DMODEL * 4);
  float*  vb      = (float*)alloc((size_t)MROWS * DMODEL * 4);
  float*  gb      = (float*)alloc((size_t)MROWS * NHEADS * 4);
  float*  betab   = (float*)alloc((size_t)MROWS * NHEADS * 4);
  float*  readsb  = (float*)alloc((size_t)MROWS * DMODEL * 4);
  __bf16* fused   = (__bf16*)alloc((size_t)MROWS * FUSEDK * 2);

  // 1. activation conversion + weight pre-packing (transpose paid once)
  int nh = MROWS * DMODEL;
  k_f2bf<<<(nh + 255) / 256, 256, 0, stream>>>(hidden, hb, nh);
  int nc = DMODEL * CTRLN;
  k_packB<<<(nc + 255) / 256, 256, 0, stream>>>(ctrl_W, ctrlWb, DMODEL, CTRLN);
  int np = DMODEL * P3N;
  k_packB<<<(np + 255) / 256, 256, 0, stream>>>(p20c_W, p20cWb, DMODEL, P3N);
  int no = FUSEDK * DMODEL;
  k_packB<<<(no + 255) / 256, 256, 0, stream>>>(out_W, outWb, FUSEDK, DMODEL);

  // 2. P20 scan
  k_p20_scan<<<(BDIM * DMODEL / 2) / 256, 256, 0, stream>>>(hidden, p20_dlog, p20_theta, p20rd);

  // 3. ctrl GEMM: [8192,1024] @ [1024,4128] + bias
  k_gemm_bf16<<<dim3((CTRLN + 127) / 128, MROWS / 128), 256, 0, stream>>>(
      hb, DMODEL, ctrlWb, CTRLN, DMODEL, ctrl, ctrl_b, nullptr, 0);

  // 4. RMSNorm(p20) -> fused[:, 1024:2048] bf16
  k_p20norm<<<MROWS, 256, 0, stream>>>(p20rd, vnorm_scale, fused);

  // 5. p20c GEMM: normed_p20 @ [1024,3072] (A read from fused, lda=2048)
  k_gemm_bf16<<<dim3(P3N / 128, MROWS / 128), 256, 0, stream>>>(
      fused + DMODEL, FUSEDK, p20cWb, P3N, DMODEL, p20qkv, nullptr, nullptr, 0);

  // 6. g / beta
  k_gbeta<<<(MROWS * NHEADS) / 256, 256, 0, stream>>>(ctrl, gb, betab);

  // 7. causal dwconv + SiLU + ramp*p20 for q,k,v
  int ncv = MROWS * DMODEL;
  k_conv<<<ncv / 256, 256, 0, stream>>>(ctrl, qconv_w, ramp_logit, p20qkv, qb, 0);
  k_conv<<<ncv / 256, 256, 0, stream>>>(ctrl, kconv_w, ramp_logit, p20qkv, kb, 1);
  k_conv<<<ncv / 256, 256, 0, stream>>>(ctrl, vconv_w, ramp_logit, p20qkv, vb, 2);

  // 8. gated delta rule scan
  k_delta<<<BDIM * NHEADS, 64, 0, stream>>>(qb, kb, vb, gb, betab, readsb);

  // 9. per-head RMSNorm -> fused[:, 0:1024] bf16
  k_mnorm<<<(MROWS * NHEADS) / 8, 256, 0, stream>>>(readsb, mnorm_scale, fused);

  // 10. out GEMM with fused bias + sigmoid(gate) epilogue -> d_out
  k_gemm_bf16<<<dim3(DMODEL / 128, MROWS / 128), 256, 0, stream>>>(
      fused, FUSEDK, outWb, DMODEL, FUSEDK, out, out_b,
      ctrl + 3 * DMODEL + 2 * NHEADS, CTRLN);

  (void)in_sizes; (void)n_in; (void)out_size; (void)ws_size;
}
